// TransformerBlock_89824946029259
// MI455X (gfx1250) — compile-verified, matrix operand order checked by reading, not run on hardware
//
#include <hip/hip_runtime.h>
#include <math.h>

#define B_ 4
#define C_ 192
#define HGT 128
#define WID 128
#define HW 16384
#define NPIX 65536          // B_*HW
#define HEADS 4
#define CH 48               // C_/HEADS
#define H2 1020             // HIDDEN*2
#define H2P 1024            // padded M for pin
#define HID 510
#define KP_POUT 512         // padded K for pout

typedef __attribute__((ext_vector_type(16))) __bf16 v16bf;
typedef __attribute__((ext_vector_type(8)))  float  v8f;
typedef __attribute__((ext_vector_type(4)))  unsigned int u32x4;
typedef __attribute__((ext_vector_type(8)))  int i32x8;
typedef __attribute__((ext_vector_type(4)))  int i32x4;

union Frag { v16bf v; __bf16 e[16]; unsigned int u[8]; };

// K indices for the 16-bit WMMA A/B fragment layout (per half-wave `hi`).
__device__ __forceinline__ void kidx(int hi, int kb[8]) {
#pragma unroll
  for (int v = 0; v < 8; ++v)
    kb[v] = (v < 4 ? 2 * v : 16 + 2 * (v - 4)) + hi * 8;
}

// ---------------- weight fp32 -> bf16 with M/K zero padding ----------------
__global__ void convert_w_bf16(const float* __restrict__ w, __bf16* __restrict__ o,
                               int M, int K, int Mpad, int Kpad) {
  size_t i = (size_t)blockIdx.x * blockDim.x + threadIdx.x;
  size_t tot = (size_t)Mpad * Kpad;
  if (i >= tot) return;
  int m = (int)(i / Kpad), k = (int)(i % Kpad);
  float v = (m < M && k < K) ? w[(size_t)m * K + k] : 0.f;
  o[i] = (__bf16)v;
}

// ---------------- LayerNorm over C (per pixel), NCHW fp32 -> [p,192] bf16 ----
__global__ void __launch_bounds__(256) layernorm_bf16(const float* __restrict__ x,
    const float* __restrict__ lw, const float* __restrict__ lb, __bf16* __restrict__ o) {
  int p = blockIdx.x * blockDim.x + threadIdx.x;   // 65536
  int bi = p >> 14, hw = p & 16383;
  const float* xp = x + (size_t)bi * C_ * HW + hw;
  float s = 0.f, s2 = 0.f;
  for (int c = 0; c < C_; ++c) { float v = xp[(size_t)c * HW]; s += v; s2 += v * v; }
  float mu = s * (1.f / C_);
  float var = s2 * (1.f / C_) - mu * mu;
  float inv = rsqrtf(var + 1e-5f);
  __bf16* op = o + (size_t)p * C_;
  for (int c = 0; c < C_; ++c) {
    float v = (xp[(size_t)c * HW] - mu) * inv * lw[c] + lb[c];
    op[c] = (__bf16)v;
  }
}

// ---------------- WMMA GEMM with TDM-staged B panel in LDS ------------------
// W: [Mpad, Kp] bf16 (zero padded rows). X: [NPIX, Kp] bf16. out/res: NCHW fp32.
// One tensor_load_to_lds per block stages X[n0_blk .. n0_blk+63][0..Kp) into LDS.
__global__ void __launch_bounds__(256) gemm_bf16_nchw(
    const __bf16* __restrict__ Wbf, const __bf16* __restrict__ Xbf,
    const float* __restrict__ bias, const float* __restrict__ res,
    float* __restrict__ out, int M, int Kp) {
  extern __shared__ __bf16 smem[];                  // [64][Kp]
  const int tid = threadIdx.x;
  const int wave = tid >> 5, lane = tid & 31;
  const int l = lane & 15, hi = lane >> 4;
  const int wm = wave >> 2, wn = wave & 3;          // 2 x 4 waves -> 32M x 64N
  const int m0 = (blockIdx.y * 2 + wm) * 16;
  const int n0 = (blockIdx.x * 4 + wn) * 16;

  // ---- TDM: DMA the 64-pixel x Kp bf16 B-panel into LDS (wave 0 issues) ----
  if (wave == 0) {
    unsigned long long ga =
        (unsigned long long)(uintptr_t)(Xbf + (size_t)blockIdx.x * 64 * Kp);
    u32x4 g0;
    g0[0] = 1u;                                        // count=1, user mode
    g0[1] = 0u;                                        // lds_addr = base of dynamic LDS
    g0[2] = (unsigned int)(ga & 0xffffffffull);        // global_addr[31:0]
    g0[3] = (unsigned int)((ga >> 32) & 0x1ffffffu)    // global_addr[56:32]
            | 0x80000000u;                             // type = 2 (bits 127:126 = 0b10)
    i32x8 g1;
    g1[0] = (int)(1u << 16);                           // data_size = 1 (2 bytes)
    g1[1] = (int)(((unsigned)Kp & 0xffffu) << 16);     // tensor_dim0[15:0] @ bits 63:48
    g1[2] = (int)((((unsigned)Kp >> 16) & 0xffffu)     // tensor_dim0[31:16]
            | (((unsigned)NPIX & 0xffffu) << 16));     // tensor_dim1[15:0]
    g1[3] = (int)((((unsigned)NPIX >> 16) & 0xffffu)   // tensor_dim1[31:16]
            | ((unsigned)Kp << 16));                   // tile_dim0 = Kp
    g1[4] = 64;                                        // tile_dim1 = 64 rows
    g1[5] = Kp;                                        // tensor_dim0_stride[31:0]
    g1[6] = 0;                                         // stride hi / dim1_stride lo
    g1[7] = 0;
    i32x4 gz4; gz4[0] = 0; gz4[1] = 0; gz4[2] = 0; gz4[3] = 0;
    i32x8 gz8;
#pragma unroll
    for (int i = 0; i < 8; ++i) gz8[i] = 0;
    __builtin_amdgcn_tensor_load_to_lds(g0, g1, gz4, gz4, gz8, 0);
    __builtin_amdgcn_s_wait_tensorcnt(0);
  }
  __syncthreads();

  int kb[8]; kidx(hi, kb);
  const __bf16* wrow = Wbf + (size_t)(m0 + l) * Kp;
  const __bf16* xrow = smem + (size_t)(wn * 16 + l) * Kp;   // B rows from LDS
  v8f acc = {};
  for (int kg = 0; kg < Kp; kg += 32) {
    Frag A, Bm;
#pragma unroll
    for (int v = 0; v < 8; ++v) {
      A.u[v]  = *(const unsigned int*)(wrow + kg + kb[v]);
      Bm.u[v] = *(const unsigned int*)(xrow + kg + kb[v]);
    }
    acc = __builtin_amdgcn_wmma_f32_16x16x32_bf16(false, A.v, false, Bm.v,
                                                  (short)0, acc, false, false);
  }
  const int n = n0 + l;
  const int bimg = n >> 14, hw = n & 16383;
#pragma unroll
  for (int r = 0; r < 8; ++r) {
    int m = m0 + hi * 8 + r;
    if (m < M) {
      size_t oi = ((size_t)bimg * M + m) * (size_t)HW + hw;
      float v = acc[r] + bias[m];
      if (res) v += res[oi];
      out[oi] = v;
    }
  }
}

// ---------------- depthwise 3x3 SAME + bias, NCHW fp32 ----------------
__global__ void __launch_bounds__(256) dwconv3x3(const float* __restrict__ in,
    const float* __restrict__ w, const float* __restrict__ bias,
    float* __restrict__ out, int Cn) {
  size_t idx = (size_t)blockIdx.x * blockDim.x + threadIdx.x;
  int hw = (int)(idx & 16383);
  size_t t = idx >> 14;                 // b*Cn + c
  int c = (int)(t % Cn);
  int row = hw >> 7, col = hw & 127;
  const float* ip = in + t * (size_t)HW;
  const float* wc = w + (size_t)c * 9;
  float acc = bias[c];
#pragma unroll
  for (int dy = -1; dy <= 1; ++dy) {
#pragma unroll
    for (int dx = -1; dx <= 1; ++dx) {
      int r = row + dy, cc = col + dx;
      if (r >= 0 && r < HGT && cc >= 0 && cc < WID)
        acc += ip[r * WID + cc] * wc[(dy + 1) * 3 + (dx + 1)];
    }
  }
  out[idx] = acc;
}

// ---------------- inverse L2 norms of q,k rows (length HW), LDS reduce -------
__global__ void __launch_bounds__(256) rownorm(const float* __restrict__ qkv,
                                               float* __restrict__ invn) {
  int r = blockIdx.x;                   // 0..1535 : (qk, b, c)
  int qk = r / (B_ * C_);
  int rem = r % (B_ * C_);
  int b = rem / C_, c = rem % C_;
  const float* row = qkv + ((size_t)b * (3 * C_) + qk * C_ + c) * HW;
  float s = 0.f;
  for (int i = threadIdx.x; i < HW; i += 256) { float v = row[i]; s += v * v; }
  __shared__ float sd[256];
  sd[threadIdx.x] = s;
  __syncthreads();
  for (int st = 128; st > 0; st >>= 1) {
    if (threadIdx.x < st) sd[threadIdx.x] += sd[threadIdx.x + st];
    __syncthreads();
  }
  if (threadIdx.x == 0) invn[r] = 1.f / fmaxf(sqrtf(sd[0]), 1e-12f);
}

// ---------------- attn = (q/|q|) . (k/|k|)^T * temp : 48x48 per (b,h) --------
__global__ void __launch_bounds__(288) attn_qk(const float* __restrict__ qkv,
    const float* __restrict__ invn, const float* __restrict__ temp,
    float* __restrict__ attn) {
  int bh = blockIdx.x, b = bh >> 2, h = bh & 3;
  int wave = threadIdx.x >> 5;          // 9 waves: 3x3 tiles
  int mt = wave / 3, nt = wave % 3;
  int lane = threadIdx.x & 31, l = lane & 15, hi = lane >> 4;
  int kb[8]; kidx(hi, kb);
  const float* qrow = qkv + ((size_t)b * (3 * C_) + h * CH + mt * 16 + l) * HW;
  const float* krow = qkv + ((size_t)b * (3 * C_) + C_ + h * CH + nt * 16 + l) * HW;
  float iq = invn[b * C_ + h * CH + mt * 16 + l];
  float ik = invn[B_ * C_ + b * C_ + h * CH + nt * 16 + l];
  v8f acc = {};
  for (int kg = 0; kg < HW; kg += 32) {
    Frag A, Bm;
#pragma unroll
    for (int v = 0; v < 8; ++v) {
      float2 fq = *(const float2*)(qrow + kg + kb[v]);
      float2 fk = *(const float2*)(krow + kg + kb[v]);
      A.e[2 * v]      = (__bf16)(fq.x * iq);
      A.e[2 * v + 1]  = (__bf16)(fq.y * iq);
      Bm.e[2 * v]     = (__bf16)(fk.x * ik);
      Bm.e[2 * v + 1] = (__bf16)(fk.y * ik);
    }
    acc = __builtin_amdgcn_wmma_f32_16x16x32_bf16(false, A.v, false, Bm.v,
                                                  (short)0, acc, false, false);
  }
  float t = temp[h];
  int n = nt * 16 + l;
#pragma unroll
  for (int r = 0; r < 8; ++r) {
    int m = mt * 16 + hi * 8 + r;
    attn[((size_t)bh * CH + m) * 64 + n] = acc[r] * t;
  }
}

// ---------------- softmax over 48, write zeros in pad cols 48..63 ------------
__global__ void softmax48(float* __restrict__ attn) {
  int r = blockIdx.x * blockDim.x + threadIdx.x;
  if (r >= B_ * HEADS * CH) return;
  float* row = attn + (size_t)r * 64;
  float mx = -3.4e38f;
  for (int i = 0; i < CH; ++i) mx = fmaxf(mx, row[i]);
  float s = 0.f;
  for (int i = 0; i < CH; ++i) { float e = expf(row[i] - mx); row[i] = e; s += e; }
  float inv = 1.f / s;
  for (int i = 0; i < CH; ++i) row[i] *= inv;
  for (int i = CH; i < 64; ++i) row[i] = 0.f;
}

// ---------------- out = attn . v  -> bf16 [p, 192] ---------------------------
__global__ void __launch_bounds__(96) attn_v(const float* __restrict__ attn,
    const float* __restrict__ qkv, __bf16* __restrict__ outbf) {
  int bh = blockIdx.y, b = bh >> 2, h = bh & 3;
  int n0 = blockIdx.x * 16;
  int wave = threadIdx.x >> 5;          // 3 waves = 3 M tiles
  int lane = threadIdx.x & 31, l = lane & 15, hi = lane >> 4;
  int kb[8]; kidx(hi, kb);
  const float* arow = attn + ((size_t)bh * CH + wave * 16 + l) * 64;
  const float* vb = qkv + ((size_t)b * (3 * C_) + 2 * C_ + h * CH) * HW + n0 + l;
  v8f acc = {};
  for (int kg = 0; kg < 64; kg += 32) {
    Frag A, Bm;
#pragma unroll
    for (int v = 0; v < 8; ++v) {
      float2 fa = *(const float2*)(arow + kg + kb[v]);
      A.e[2 * v]     = (__bf16)fa.x;
      A.e[2 * v + 1] = (__bf16)fa.y;
      int k0 = kg + kb[v];
      float b0 = (k0 < CH)     ? vb[(size_t)k0 * HW]       : 0.f;
      float b1 = (k0 + 1 < CH) ? vb[(size_t)(k0 + 1) * HW] : 0.f;
      Bm.e[2 * v]     = (__bf16)b0;
      Bm.e[2 * v + 1] = (__bf16)b1;
    }
    acc = __builtin_amdgcn_wmma_f32_16x16x32_bf16(false, A.v, false, Bm.v,
                                                  (short)0, acc, false, false);
  }
  size_t pix = (size_t)b * HW + n0 + l;
#pragma unroll
  for (int r = 0; r < 8; ++r) {
    int m = wave * 16 + hi * 8 + r;      // 0..47 channel within head
    outbf[pix * C_ + h * CH + m] = (__bf16)acc[r];
  }
}

// ---------------- GLU: gelu(x1)*x2 -> bf16 [p, 512] (pad 510,511 = 0) --------
__global__ void __launch_bounds__(256) glu_bf16(const float* __restrict__ dw,
                                                __bf16* __restrict__ o) {
  int p = blockIdx.x * blockDim.x + threadIdx.x;
  if (p >= NPIX) return;
  int b = p >> 14, hw = p & 16383;
  const float* base = dw + (size_t)b * H2 * HW + hw;
  __bf16* op = o + (size_t)p * KP_POUT;
  for (int c = 0; c < HID; ++c) {
    float x1 = base[(size_t)c * HW];
    float x2 = base[(size_t)(c + HID) * HW];
    float g = 0.5f * x1 * (1.f + erff(x1 * 0.7071067811865476f));
    op[c] = (__bf16)(g * x2);
  }
  op[510] = (__bf16)0.f;
  op[511] = (__bf16)0.f;
}

extern "C" void kernel_launch(void* const* d_in, const int* in_sizes, int n_in,
                              void* d_out, int out_size, void* d_ws, size_t ws_size,
                              hipStream_t stream) {
  const float* x      = (const float*)d_in[0];
  const float* ln1_w  = (const float*)d_in[1];
  const float* ln1_b  = (const float*)d_in[2];
  const float* qkv_w  = (const float*)d_in[3];
  const float* qkv_b  = (const float*)d_in[4];
  const float* qdw_w  = (const float*)d_in[5];
  const float* qdw_b  = (const float*)d_in[6];
  const float* temp   = (const float*)d_in[7];
  const float* proj_w = (const float*)d_in[8];
  const float* proj_b = (const float*)d_in[9];
  const float* ln2_w  = (const float*)d_in[10];
  const float* ln2_b  = (const float*)d_in[11];
  const float* pin_w  = (const float*)d_in[12];
  const float* pin_b  = (const float*)d_in[13];
  const float* dw_w   = (const float*)d_in[14];
  const float* dw_b   = (const float*)d_in[15];
  const float* pout_w = (const float*)d_in[16];
  const float* pout_b = (const float*)d_in[17];
  float* out = (float*)d_out;

  // ---- workspace layout ----
  char* ws = (char*)d_ws;
  size_t off = 0;
  auto alloc = [&](size_t bytes) { char* p = ws + off; off = (off + bytes + 255) & ~(size_t)255; return p; };
  __bf16* wqkvBf  = (__bf16*)alloc((size_t)3 * C_ * C_ * 2);           // 576x192
  __bf16* wprojBf = (__bf16*)alloc((size_t)C_ * C_ * 2);               // 192x192
  __bf16* wpinBf  = (__bf16*)alloc((size_t)H2P * C_ * 2);              // 1024x192
  __bf16* wpoutBf = (__bf16*)alloc((size_t)C_ * KP_POUT * 2);          // 192x512
  float*  invn    = (float*)alloc((size_t)2 * B_ * C_ * 4);            // 1536
  float*  attnB   = (float*)alloc((size_t)B_ * HEADS * CH * 64 * 4);   // padded logits
  __bf16* xlnBf   = (__bf16*)alloc((size_t)NPIX * C_ * 2);             // LN1 / LN2 (reused)
  __bf16* aoutBf  = (__bf16*)alloc((size_t)NPIX * C_ * 2);             // attn output
  float*  x1      = (float*)alloc((size_t)B_ * C_ * HW * 4);           // after attention residual
  __bf16* gluBf   = (__bf16*)alloc((size_t)NPIX * KP_POUT * 2);
  float*  bufA    = (float*)alloc((size_t)B_ * H2 * HW * 4);           // qkv GEMM out / pin out
  float*  bufB    = (float*)alloc((size_t)B_ * H2 * HW * 4);           // post-dwconv
  (void)ws_size; (void)in_sizes; (void)n_in; (void)out_size;

  // ---- weight conversion ----
  {
    size_t t;
    t = (size_t)3 * C_ * C_;
    convert_w_bf16<<<(unsigned)((t + 255) / 256), 256, 0, stream>>>(qkv_w, wqkvBf, 3 * C_, C_, 3 * C_, C_);
    t = (size_t)C_ * C_;
    convert_w_bf16<<<(unsigned)((t + 255) / 256), 256, 0, stream>>>(proj_w, wprojBf, C_, C_, C_, C_);
    t = (size_t)H2P * C_;
    convert_w_bf16<<<(unsigned)((t + 255) / 256), 256, 0, stream>>>(pin_w, wpinBf, H2, C_, H2P, C_);
    t = (size_t)C_ * KP_POUT;
    convert_w_bf16<<<(unsigned)((t + 255) / 256), 256, 0, stream>>>(pout_w, wpoutBf, C_, HID, C_, KP_POUT);
  }

  // ---- attention branch ----
  layernorm_bf16<<<NPIX / 256, 256, 0, stream>>>(x, ln1_w, ln1_b, xlnBf);
  gemm_bf16_nchw<<<dim3(NPIX / 64, (3 * C_) / 32), 256, (size_t)C_ * 64 * 2, stream>>>(
      wqkvBf, xlnBf, qkv_b, nullptr, bufA, 3 * C_, C_);
  dwconv3x3<<<(unsigned)((size_t)B_ * 3 * C_ * HW / 256), 256, 0, stream>>>(
      bufA, qdw_w, qdw_b, bufB, 3 * C_);
  rownorm<<<2 * B_ * C_, 256, 0, stream>>>(bufB, invn);
  attn_qk<<<B_ * HEADS, 288, 0, stream>>>(bufB, invn, temp, attnB);
  softmax48<<<3, 256, 0, stream>>>(attnB);
  attn_v<<<dim3(HW / 16, B_ * HEADS), 96, 0, stream>>>(attnB, bufB, aoutBf);
  gemm_bf16_nchw<<<dim3(NPIX / 64, C_ / 32), 256, (size_t)C_ * 64 * 2, stream>>>(
      wprojBf, aoutBf, proj_b, x, x1, C_, C_);

  // ---- FFN branch ----
  layernorm_bf16<<<NPIX / 256, 256, 0, stream>>>(x1, ln2_w, ln2_b, xlnBf);
  gemm_bf16_nchw<<<dim3(NPIX / 64, H2P / 32), 256, (size_t)C_ * 64 * 2, stream>>>(
      wpinBf, xlnBf, pin_b, nullptr, bufA, H2, C_);
  dwconv3x3<<<(unsigned)((size_t)B_ * H2 * HW / 256), 256, 0, stream>>>(
      bufA, dw_w, dw_b, bufB, H2);
  glu_bf16<<<NPIX / 256, 256, 0, stream>>>(bufB, gluBf);
  gemm_bf16_nchw<<<dim3(NPIX / 64, C_ / 32), 256, (size_t)KP_POUT * 64 * 2, stream>>>(
      wpoutBf, gluBf, pout_b, x1, out, C_, KP_POUT);
}
